// RecognitionModule_34067680592194
// MI455X (gfx1250) — matrix operand order repaired and checked by reading, not compiled, not run on hardware
//
#include <hip/hip_runtime.h>
#include <hip/hip_bf16.h>
#include <math.h>

// ---- problem constants (match reference) ----
#define LNUM 3
#define HD   1024
#define NHD  8
#define DHD  128
#define ED   768
#define FFD  4096
#define ADIM 100
#define IDIM 50
#define BD   4
#define SD   1024
#define MTOT (BD * SD)     // 4096 rows of activations
#define LN_EPS 1e-5f

typedef __bf16 bf16_t;
typedef __attribute__((ext_vector_type(16))) __bf16 v16bf;
typedef __attribute__((ext_vector_type(8)))  float  v8f;

__device__ __forceinline__ float gelu_exact(float x) {
  return 0.5f * x * (1.0f + erff(x * 0.70710678118654752440f));
}

// ---------------------------------------------------------------------------
// f32 -> bf16 conversion (weights, done once per launch)
// ---------------------------------------------------------------------------
__global__ void cvt_bf16_k(const float* __restrict__ src, bf16_t* __restrict__ dst,
                           long long n) {
  long long i = (long long)blockIdx.x * blockDim.x + threadIdx.x;
  if (i < n) dst[i] = (bf16_t)src[i];
}

// ---------------------------------------------------------------------------
// Generic batched GEMM:  C = act(A[M,K] * W[N,K]^T + bias)
//   A: f32 row-major (lda, batch stride sA) -> converted to bf16 in-register
//   W: bf16 row-major [N,K] (ldb, batch stride sB)
//   C: f32 (ldc), batch offset = (z/cDiv)*sC1 + (z%cDiv)*sC2
// One wave -> 32(M) x 64(N) tile: 2 A-frags, 4 B-frags, 8 WMMAs per K=32 step.
// Block of 8 waves covers 128(M) x 128(N).  Branch-free inner loop: weight row
// indices are clamped to N-1 up front (loads legal), dead columns are simply
// not stored in the epilogue.
// Requires: M % 32 == 0, K % 32 == 0 (true for every use here).
// ---------------------------------------------------------------------------
__global__ __launch_bounds__(256)
void gemm_bf16_wmma(const float* __restrict__ Ag, const bf16_t* __restrict__ Wg,
                    const float* __restrict__ bias, float* __restrict__ Cg,
                    int M, int N, int K, int lda, int ldb, int ldc,
                    long long sA, long long sB, long long sC1, long long sC2,
                    int cDiv, int act) {
  const int z = blockIdx.z;
  const float*  Ab = Ag + (long long)z * sA;
  const bf16_t* Wb = Wg + (long long)z * sB;
  float*        Cb = Cg + (long long)(z / cDiv) * sC1 + (long long)(z % cDiv) * sC2;

  const int lane = threadIdx.x & 31;
  const int wv   = threadIdx.x >> 5;                   // 0..7
  const int m0   = blockIdx.y * 128 + (wv >> 1) * 32;  // 4 waves along M
  const int n0   = blockIdx.x * 128 + (wv & 1) * 64;   // 2 waves along N
  if (m0 >= M) return;

  const int half = lane >> 4;   // 0: lanes 0-15, 1: lanes 16-31
  const int lr   = lane & 15;   // row (A) / col (B) within a 16-wide tile

  // Pre-offset row pointers; per ISA layout the per-lane A data is two
  // contiguous 8-float runs at [k0 + 8*half] and [k0 + 16 + 8*half].
  const float* arow0 = Ab + (long long)(m0 + lr) * lda + (half << 3);
  const float* arow1 = arow0 + (long long)16 * lda;

  const bf16_t* brow[4];
  int ncol[4];
#pragma unroll
  for (int t = 0; t < 4; ++t) {
    const int n = n0 + t * 16 + lr;
    ncol[t] = n;
    const int nc = (n < N) ? n : (N - 1);               // clamp: loads stay legal
    brow[t] = Wb + (long long)nc * ldb + (half << 3);   // two 8-bf16 runs at +0 and +16
  }

  v8f acc[8];  // [mi*4 + t]
#pragma unroll
  for (int i = 0; i < 8; ++i)
#pragma unroll
    for (int v = 0; v < 8; ++v) acc[i][v] = 0.0f;

  for (int k0 = 0; k0 < K; k0 += 32) {
    const float* pa0 = arow0 + k0;
    const float* pa1 = arow1 + k0;
    __builtin_prefetch(pa0 + 32, 0, 1);   // global_prefetch_b8 for next K-step
    __builtin_prefetch(pa1 + 32, 0, 1);

    v16bf a0, a1;
#pragma unroll
    for (int e = 0; e < 8; ++e) {
      a0[e]     = (bf16_t)pa0[e];
      a0[e + 8] = (bf16_t)pa0[e + 16];
      a1[e]     = (bf16_t)pa1[e];
      a1[e + 8] = (bf16_t)pa1[e + 16];
    }
#pragma unroll
    for (int t = 0; t < 4; ++t) {
      const bf16_t* pb = brow[t] + k0;
      v16bf bfr;
#pragma unroll
      for (int e = 0; e < 16; ++e) bfr[e] = pb[e + ((e >> 3) << 3)];
      acc[t] = __builtin_amdgcn_wmma_f32_16x16x32_bf16(
          false, a0, false, bfr, (short)0, acc[t], false, false);
      acc[4 + t] = __builtin_amdgcn_wmma_f32_16x16x32_bf16(
          false, a1, false, bfr, (short)0, acc[4 + t], false, false);
    }
  }

  // Epilogue: C/D layout -> VGPR v holds row (base + v + 8*half), column ncol[t]
#pragma unroll
  for (int t = 0; t < 4; ++t) {
    const int n = ncol[t];
    if (n >= N) continue;
    const float bv = bias ? bias[n] : 0.0f;
#pragma unroll
    for (int mi = 0; mi < 2; ++mi) {
      const int mbase = m0 + mi * 16 + (half << 3);
#pragma unroll
      for (int v = 0; v < 8; ++v) {
        float x = acc[mi * 4 + t][v] + bv;
        if (act == 1) x = gelu_exact(x);
        Cb[(long long)(mbase + v) * ldc + n] = x;
      }
    }
  }
}

// ---------------------------------------------------------------------------
// h += positional encoding, then LayerNorm(ln_in).  One block (256 thr) per row.
// ---------------------------------------------------------------------------
__global__ __launch_bounds__(256)
void addpos_ln_k(float* __restrict__ h, const float* __restrict__ g,
                 const float* __restrict__ b) {
  const int m = blockIdx.x;
  const int s = m & (SD - 1);
  const int tid = threadIdx.x;
  __shared__ float r1[256], r2[256];
  float vals[4];
  float sum = 0.f, sq = 0.f;
  const float nl = -logf(10000.0f) / (float)HD;
#pragma unroll
  for (int j = 0; j < 4; ++j) {
    const int c = tid + j * 256;
    const float div = __expf(nl * (float)((c >> 1) << 1));
    const float ang = (float)s * div;
    const float pe = (c & 1) ? cosf(ang) : sinf(ang);
    const float v = h[(long long)m * HD + c] + pe;
    vals[j] = v; sum += v; sq += v * v;
  }
  r1[tid] = sum; r2[tid] = sq; __syncthreads();
  for (int st = 128; st > 0; st >>= 1) {
    if (tid < st) { r1[tid] += r1[tid + st]; r2[tid] += r2[tid + st]; }
    __syncthreads();
  }
  const float mean = r1[0] * (1.0f / HD);
  const float var  = r2[0] * (1.0f / HD) - mean * mean;
  const float inv  = rsqrtf(var + LN_EPS);
#pragma unroll
  for (int j = 0; j < 4; ++j) {
    const int c = tid + j * 256;
    h[(long long)m * HD + c] = (vals[j] - mean) * inv * g[c] + b[c];
  }
}

// ---------------------------------------------------------------------------
// dst = LayerNorm(x + y) ; y may be null (plain LN). Block per row (H=1024).
// ---------------------------------------------------------------------------
__global__ __launch_bounds__(256)
void resid_ln_k(const float* __restrict__ x, const float* __restrict__ y,
                const float* __restrict__ g, const float* __restrict__ b,
                float* __restrict__ dst) {
  const int m = blockIdx.x;
  const int tid = threadIdx.x;
  __shared__ float r1[256], r2[256];
  float vals[4];
  float sum = 0.f, sq = 0.f;
#pragma unroll
  for (int j = 0; j < 4; ++j) {
    const int c = tid + j * 256;
    float v = x[(long long)m * HD + c];
    if (y) v += y[(long long)m * HD + c];
    vals[j] = v; sum += v; sq += v * v;
  }
  r1[tid] = sum; r2[tid] = sq; __syncthreads();
  for (int st = 128; st > 0; st >>= 1) {
    if (tid < st) { r1[tid] += r1[tid + st]; r2[tid] += r2[tid + st]; }
    __syncthreads();
  }
  const float mean = r1[0] * (1.0f / HD);
  const float var  = r2[0] * (1.0f / HD) - mean * mean;
  const float inv  = rsqrtf(var + LN_EPS);
#pragma unroll
  for (int j = 0; j < 4; ++j) {
    const int c = tid + j * 256;
    dst[(long long)m * HD + c] = (vals[j] - mean) * inv * g[c] + b[c];
  }
}

// ---------------------------------------------------------------------------
// Split packed qkv [M, 3H] into head-major Q(f32), K(bf16) [B,NH,S,DH]
// and V transposed (bf16) [B,NH,DH,S] so the attn*V GEMM is A*W^T form.
// ---------------------------------------------------------------------------
__global__ void pack_qkv_k(const float* __restrict__ qkv, float* __restrict__ Q,
                           bf16_t* __restrict__ Kh, bf16_t* __restrict__ Vt) {
  long long i = (long long)blockIdx.x * blockDim.x + threadIdx.x;
  if (i >= (long long)MTOT * HD) return;
  const int m = (int)(i / HD), c = (int)(i % HD);
  const int bb = m / SD, s = m & (SD - 1);
  const int hh = c / DHD, d = c & (DHD - 1);
  const float* row = qkv + (long long)m * (3 * HD);
  const long long bh = (long long)bb * NHD + hh;
  Q [ (bh * SD + s) * DHD + d] = row[c];
  Kh[ (bh * SD + s) * DHD + d] = (bf16_t)row[HD + c];
  Vt[ (bh * DHD + d) * SD + s] = (bf16_t)row[2 * HD + c];
}

// ---------------------------------------------------------------------------
// Row softmax in place: softmax(scale * x) over rowLen (multiple of 256).
// ---------------------------------------------------------------------------
__global__ __launch_bounds__(256)
void softmax_rows_k(float* __restrict__ x, int rowLen, float scale) {
  float* row = x + (long long)blockIdx.x * rowLen;
  const int tid = threadIdx.x;
  __shared__ float red[256];
  float mx = -3.402823e38f;
  for (int c = tid; c < rowLen; c += 256) mx = fmaxf(mx, row[c]);
  red[tid] = mx; __syncthreads();
  for (int st = 128; st > 0; st >>= 1) {
    if (tid < st) red[tid] = fmaxf(red[tid], red[tid + st]);
    __syncthreads();
  }
  mx = red[0]; __syncthreads();
  float sum = 0.f;
  for (int c = tid; c < rowLen; c += 256) {
    const float e = __expf(scale * (row[c] - mx));
    row[c] = e; sum += e;
  }
  red[tid] = sum; __syncthreads();
  for (int st = 128; st > 0; st >>= 1) {
    if (tid < st) red[tid] += red[tid + st];
    __syncthreads();
  }
  const float inv = 1.0f / red[0];
  for (int c = tid; c < rowLen; c += 256) row[c] *= inv;
}

// out[m] = dot(A row m, w) + bias[0] ; mode 1 -> sigmoid.  Block per row.
__global__ __launch_bounds__(256)
void rowdot_k(const float* __restrict__ Ag, int K, const float* __restrict__ w,
              const float* __restrict__ bias, float* __restrict__ out, int mode) {
  const int m = blockIdx.x;
  const int tid = threadIdx.x;
  __shared__ float red[256];
  const float* r = Ag + (long long)m * K;
  float s = 0.f;
  for (int k = tid; k < K; k += 256) s += r[k] * w[k];
  red[tid] = s; __syncthreads();
  for (int st = 128; st > 0; st >>= 1) {
    if (tid < st) red[tid] += red[tid + st];
    __syncthreads();
  }
  if (tid == 0) {
    float v = red[0] + bias[0];
    if (mode == 1) v = 1.0f / (1.0f + __expf(-v));
    out[m] = v;
  }
}

__global__ void copy_f32_k(const float* __restrict__ s, float* __restrict__ d,
                           long long n) {
  long long i = (long long)blockIdx.x * blockDim.x + threadIdx.x;
  if (i < n) d[i] = s[i];
}

// global_ctx[b, c] = sum_s pool[b,s] * ctx[b,s,c]
__global__ void global_ctx_k(const float* __restrict__ pool,
                             const float* __restrict__ ctx,
                             float* __restrict__ out) {
  const int i = blockIdx.x * blockDim.x + threadIdx.x;
  if (i >= BD * HD) return;
  const int b = i / HD, c = i % HD;
  float s = 0.f;
  for (int t = 0; t < SD; ++t)
    s += pool[b * SD + t] * ctx[((long long)b * SD + t) * HD + c];
  out[i] = s;
}

// EMA smoothing along sequence for the concatenated [action|instr] channels.
__global__ void ema_k(const float* __restrict__ act, const float* __restrict__ ins,
                      const float* __restrict__ sfp, float* __restrict__ outAct,
                      float* __restrict__ outIns) {
  const int i = blockIdx.x * blockDim.x + threadIdx.x;
  if (i >= BD * (ADIM + IDIM)) return;
  const int b = i / (ADIM + IDIM);
  const int c = i % (ADIM + IDIM);
  const float sf = fminf(fmaxf(sfp[0], 0.0f), 1.0f);
  const float* src; float* dst; int stride;
  if (c < ADIM) {
    src = act + (long long)b * SD * ADIM + c;
    dst = outAct + (long long)b * SD * ADIM + c;
    stride = ADIM;
  } else {
    src = ins + (long long)b * SD * IDIM + (c - ADIM);
    dst = outIns + (long long)b * SD * IDIM + (c - ADIM);
    stride = IDIM;
  }
  float carry = src[0];
  dst[0] = carry;
  for (int t = 1; t < SD; ++t) {
    const float x = src[(long long)t * stride];
    carry = sf * x + (1.0f - sf) * carry;
    dst[(long long)t * stride] = carry;
  }
}

// ---------------------------------------------------------------------------
extern "C" void kernel_launch(void* const* d_in, const int* in_sizes, int n_in,
                              void* d_out, int out_size, void* d_ws, size_t ws_size,
                              hipStream_t stream) {
  (void)in_sizes; (void)n_in; (void)out_size; (void)ws_size;

  const float* frame   = (const float*)d_in[0];
  const float* W_in    = (const float*)d_in[1];
  const float* b_in    = (const float*)d_in[2];
  const float* ln_in_g = (const float*)d_in[3];
  const float* ln_in_b = (const float*)d_in[4];
  const float* qkv_w   = (const float*)d_in[5];
  const float* qkv_b   = (const float*)d_in[6];
  const float* out_w   = (const float*)d_in[7];
  const float* out_b   = (const float*)d_in[8];
  const float* ff1_w   = (const float*)d_in[9];
  const float* ff1_b   = (const float*)d_in[10];
  const float* ff2_w   = (const float*)d_in[11];
  const float* ff2_b   = (const float*)d_in[12];
  const float* n1_g    = (const float*)d_in[13];
  const float* n1_b    = (const float*)d_in[14];
  const float* n2_g    = (const float*)d_in[15];
  const float* n2_b    = (const float*)d_in[16];
  const float* lo_g    = (const float*)d_in[17];
  const float* lo_b    = (const float*)d_in[18];
  const float* a1_w    = (const float*)d_in[19];
  const float* a1_b    = (const float*)d_in[20];
  const float* a2_w    = (const float*)d_in[21];
  const float* a2_b    = (const float*)d_in[22];
  const float* i1_w    = (const float*)d_in[23];
  const float* i1_b    = (const float*)d_in[24];
  const float* i2_w    = (const float*)d_in[25];
  const float* i2_b    = (const float*)d_in[26];
  const float* c1_w    = (const float*)d_in[27];
  const float* c1_b    = (const float*)d_in[28];
  const float* c2_w    = (const float*)d_in[29];
  const float* c2_b    = (const float*)d_in[30];
  const float* p_w     = (const float*)d_in[31];
  const float* p_b     = (const float*)d_in[32];
  const float* sf      = (const float*)d_in[33];

  float* out        = (float*)d_out;
  float* out_action = out;                                   // [B,S,A]
  float* out_instr  = out_action + (size_t)MTOT * ADIM;      // [B,S,I]
  float* out_conf   = out_instr + (size_t)MTOT * IDIM;       // [B,S,1]
  float* out_pool   = out_conf + MTOT;                       // [B,S]
  float* out_gctx   = out_pool + MTOT;                       // [B,H]

  // ---- workspace carve ----
  char* ws = (char*)d_ws;
  size_t off = 0;
  auto carve = [&](size_t bytes) -> void* {
    off = (off + 255) & ~(size_t)255;
    void* p = ws + off;
    off += bytes;
    return p;
  };
  bf16_t* wbin = (bf16_t*)carve((size_t)HD * ED * 2);
  bf16_t* wqkv = (bf16_t*)carve((size_t)LNUM * 3 * HD * HD * 2);
  bf16_t* wout = (bf16_t*)carve((size_t)LNUM * HD * HD * 2);
  bf16_t* wff1 = (bf16_t*)carve((size_t)LNUM * FFD * HD * 2);
  bf16_t* wff2 = (bf16_t*)carve((size_t)LNUM * HD * FFD * 2);
  bf16_t* wa1  = (bf16_t*)carve((size_t)HD * HD * 2);
  bf16_t* wa2  = (bf16_t*)carve((size_t)ADIM * HD * 2);
  bf16_t* wi1  = (bf16_t*)carve((size_t)HD * HD * 2);
  bf16_t* wi2  = (bf16_t*)carve((size_t)IDIM * HD * 2);
  bf16_t* wc1  = (bf16_t*)carve((size_t)(HD / 2) * HD * 2);
  float* hbuf   = (float*)carve((size_t)MTOT * HD * 4);
  float* ybuf   = (float*)carve((size_t)MTOT * HD * 4);
  float* qkvbuf = (float*)carve((size_t)MTOT * 3 * HD * 4);   // reused as ctx
  float* Qh     = (float*)carve((size_t)MTOT * HD * 4);
  bf16_t* Kh    = (bf16_t*)carve((size_t)MTOT * HD * 2);
  bf16_t* Vt    = (bf16_t*)carve((size_t)MTOT * HD * 2);
  float* scores = (float*)carve((size_t)BD * NHD * SD * SD * 4);
  float* ffbuf  = (float*)carve((size_t)MTOT * FFD * 4);      // also head tmp
  float* ctxF   = (float*)carve((size_t)MTOT * HD * 4);
  float* araw   = (float*)carve((size_t)MTOT * ADIM * 4);
  float* iraw   = (float*)carve((size_t)MTOT * IDIM * 4);
  float* plog   = (float*)carve((size_t)MTOT * 4);

  auto cvt = [&](const float* s, bf16_t* d, long long n) {
    cvt_bf16_k<<<dim3((unsigned)((n + 255) / 256)), dim3(256), 0, stream>>>(s, d, n);
  };
  auto gemm = [&](const float* Ap, const bf16_t* Wp, const float* bp, float* Cp,
                  int M, int N, int K, int lda, int ldb, int ldc,
                  long long sA, long long sB, long long sC1, long long sC2,
                  int cDiv, int batch, int act) {
    dim3 g((N + 127) / 128, (M + 127) / 128, batch);
    gemm_bf16_wmma<<<g, dim3(256), 0, stream>>>(Ap, Wp, bp, Cp, M, N, K, lda, ldb,
                                                ldc, sA, sB, sC1, sC2, cDiv, act);
  };

  // ---- weight conversions (f32 -> bf16) ----
  cvt(W_in, wbin, (long long)HD * ED);
  cvt(qkv_w, wqkv, (long long)LNUM * 3 * HD * HD);
  cvt(out_w, wout, (long long)LNUM * HD * HD);
  cvt(ff1_w, wff1, (long long)LNUM * FFD * HD);
  cvt(ff2_w, wff2, (long long)LNUM * HD * FFD);
  cvt(a1_w, wa1, (long long)HD * HD);
  cvt(a2_w, wa2, (long long)ADIM * HD);
  cvt(i1_w, wi1, (long long)HD * HD);
  cvt(i2_w, wi2, (long long)IDIM * HD);
  cvt(c1_w, wc1, (long long)(HD / 2) * HD);

  // ---- input projection + positional encoding + LN ----
  gemm(frame, wbin, b_in, hbuf, MTOT, HD, ED, ED, ED, HD, 0, 0, 0, 0, 1, 1, 0);
  addpos_ln_k<<<MTOT, 256, 0, stream>>>(hbuf, ln_in_g, ln_in_b);

  const float attn_scale = 1.0f / sqrtf((float)DHD);

  for (int l = 0; l < LNUM; ++l) {
    // QKV
    gemm(hbuf, wqkv + (size_t)l * 3 * HD * HD, qkv_b + (size_t)l * 3 * HD, qkvbuf,
         MTOT, 3 * HD, HD, HD, HD, 3 * HD, 0, 0, 0, 0, 1, 1, 0);
    pack_qkv_k<<<dim3((MTOT * HD) / 256), 256, 0, stream>>>(qkvbuf, Qh, Kh, Vt);
    // scores = Q @ K^T per (b,h)
    gemm(Qh, Kh, nullptr, scores, SD, SD, DHD, DHD, DHD, SD,
         (long long)SD * DHD, (long long)SD * DHD, (long long)SD * SD, 0, 1,
         BD * NHD, 0);
    softmax_rows_k<<<BD * NHD * SD, 256, 0, stream>>>(scores, SD, attn_scale);
    // ctx = attn @ V  (V stored transposed) -> head-interleaved [M,H] (reuse qkvbuf)
    gemm(scores, Vt, nullptr, qkvbuf, SD, DHD, SD, SD, SD, HD,
         (long long)SD * SD, (long long)DHD * SD, (long long)SD * HD, (long long)DHD,
         NHD, BD * NHD, 0);
    // out projection
    gemm(qkvbuf, wout + (size_t)l * HD * HD, out_b + (size_t)l * HD, ybuf,
         MTOT, HD, HD, HD, HD, HD, 0, 0, 0, 0, 1, 1, 0);
    resid_ln_k<<<MTOT, 256, 0, stream>>>(hbuf, ybuf, n1_g + (size_t)l * HD,
                                         n1_b + (size_t)l * HD, hbuf);
    // FFN
    gemm(hbuf, wff1 + (size_t)l * FFD * HD, ff1_b + (size_t)l * FFD, ffbuf,
         MTOT, FFD, HD, HD, HD, FFD, 0, 0, 0, 0, 1, 1, 1);
    gemm(ffbuf, wff2 + (size_t)l * HD * FFD, ff2_b + (size_t)l * HD, ybuf,
         MTOT, HD, FFD, FFD, FFD, HD, 0, 0, 0, 0, 1, 1, 0);
    resid_ln_k<<<MTOT, 256, 0, stream>>>(hbuf, ybuf, n2_g + (size_t)l * HD,
                                         n2_b + (size_t)l * HD, hbuf);
  }

  // final LN -> ctxF
  resid_ln_k<<<MTOT, 256, 0, stream>>>(hbuf, nullptr, lo_g, lo_b, ctxF);

  // heads
  gemm(ctxF, wa1, a1_b, ffbuf, MTOT, HD, HD, HD, HD, HD, 0, 0, 0, 0, 1, 1, 1);
  gemm(ffbuf, wa2, a2_b, araw, MTOT, ADIM, HD, HD, HD, ADIM, 0, 0, 0, 0, 1, 1, 0);
  gemm(ctxF, wi1, i1_b, ffbuf, MTOT, HD, HD, HD, HD, HD, 0, 0, 0, 0, 1, 1, 1);
  gemm(ffbuf, wi2, i2_b, iraw, MTOT, IDIM, HD, HD, HD, IDIM, 0, 0, 0, 0, 1, 1, 0);
  gemm(ctxF, wc1, c1_b, ffbuf, MTOT, HD / 2, HD, HD, HD, HD / 2, 0, 0, 0, 0, 1, 1, 1);
  rowdot_k<<<MTOT, 256, 0, stream>>>(ffbuf, HD / 2, c2_w, c2_b, out_conf, 1);

  // pooling: softmax over sequence, weighted sum of ctx
  rowdot_k<<<MTOT, 256, 0, stream>>>(ctxF, HD, p_w, p_b, plog, 0);
  softmax_rows_k<<<BD, 256, 0, stream>>>(plog, SD, 1.0f);
  copy_f32_k<<<(MTOT + 255) / 256, 256, 0, stream>>>(plog, out_pool, MTOT);
  global_ctx_k<<<(BD * HD + 255) / 256, 256, 0, stream>>>(plog, ctxF, out_gctx);

  // temporal EMA smoothing of action/instr
  ema_k<<<(BD * (ADIM + IDIM) + 255) / 256, 256, 0, stream>>>(araw, iraw, sf,
                                                              out_action, out_instr);
}
